// LinearAttention_15891378995472
// MI455X (gfx1250) — matrix operand order repaired and checked by reading, not compiled, not run on hardware
//
#include <hip/hip_runtime.h>

// Problem constants (B, H, L, D) = (4, 16, 8192, 64)
#define BB 4
#define HH 16
#define LL 8192
#define DD 64

#define SUB 128   // rows staged per LDS refill (pass 1)
#define NSUB 4    // sub-chunks per block -> 512 rows / block in pass 1
#define ROWS 128  // rows per block in pass 2

typedef __attribute__((ext_vector_type(16))) __bf16 v16bf;
typedef __attribute__((ext_vector_type(8)))  __bf16 v8bf;
typedef __attribute__((ext_vector_type(8)))  float  v8f;

__device__ __forceinline__ unsigned short f32_to_bf16_bits(float f) {
  unsigned u = __float_as_uint(f);
  u += 0x7FFFu + ((u >> 16) & 1u);  // round-to-nearest-even
  return (unsigned short)(u >> 16);
}
__device__ __forceinline__ float elup1(float x) {  // elu(x) + 1
  return x > 0.0f ? x + 1.0f : __expf(x);
}
// Assemble a 16-element bf16 fragment from two contiguous 16B LDS halves.
// Fragment storage: [half(2)][lane(32)][8 bf16] -> two ds_load_b128 per lane.
__device__ __forceinline__ v16bf ldfrag(const v8bf* __restrict__ base, int lane) {
  v8bf lo = base[lane];
  v8bf hi = base[32 + lane];
  return __builtin_shufflevector(lo, hi, 0, 1, 2, 3, 4, 5, 6, 7,
                                 8, 9, 10, 11, 12, 13, 14, 15);
}

// s = L^{-1/4} for L = 8192
#define SCALE 0.10511205190671431f

__global__ void zero_kv_kernel(float* __restrict__ kv, int n) {
  int i = blockIdx.x * blockDim.x + threadIdx.x;
  if (i < n) kv[i] = 0.0f;
}

// -------- Pass 1: KV[bh] += Kf^T @ Vm over a 512-row slice of L ------------
// Fragment-order LDS: sKf frag f=(kk*4+dt), sVf frag g=(kk*4+et); 64 v8bf each.
__global__ __launch_bounds__(256) void kv_kernel(
    const float* __restrict__ K, const float* __restrict__ V,
    const float* __restrict__ mask, const float* __restrict__ pi,
    const float* __restrict__ mu, float* __restrict__ KV) {
  __shared__ v8bf sKf[16 * 64];  // 16 KB
  __shared__ v8bf sVf[16 * 64];  // 16 KB
  __shared__ float sMoff[DD];
  unsigned short* sKfu = (unsigned short*)sKf;
  unsigned short* sVfu = (unsigned short*)sVf;

  const int bh = blockIdx.y;  // 0..63
  const int b = bh / HH, h = bh % HH;
  const int lbase = blockIdx.x * (SUB * NSUB);
  const int tid = threadIdx.x;
  const int wave = tid >> 5;
  const int lane = tid & 31;
  const int hi = (lane >> 4) & 1;
  const int n0 = lane & 15;

  const float p0 = fminf(fmaxf(pi[0], 0.0f), 1.0f);
  const float p1 = fminf(fmaxf(pi[1], 0.0f), 1.0f);
  const float psum = p0 + p1;
  if (tid < DD)
    sMoff[tid] = p0 * mu[h * DD + tid] + p1 * mu[HH * DD + h * DD + tid];

  // Wave w owns KV tiles t=2w,2w+1: same A (dt), two B (et0, et1).
  const int dt = wave >> 1;
  const int et0 = (wave * 2) & 3;
  const int et1 = (wave * 2 + 1) & 3;
  v8f acc0 = {};
  v8f acc1 = {};

  for (int sc = 0; sc < NSUB; ++sc) {
    __syncthreads();  // previous compute done (sMoff visible on sc==0)
    const int l0 = lbase + sc * SUB;
    // Stage SUB x 64 of Kf / Vm as bf16 directly in fragment order.
    for (int j = tid; j < SUB * (DD / 4); j += 256) {
      const int row = j >> 4;
      const int col = (j & 15) * 4;
      const int l = l0 + row;
      const int kk = row >> 5;     // K=32 chunk
      const int k = row & 31;      // K index within chunk
      const float m = mask[b * LL + l];
      const float4 kq = *(const float4*)(K + ((size_t)bh * LL + l) * DD + col);
      const float4 vq = *(const float4*)(V + ((size_t)bh * LL + l) * DD + col);
      const float kvals[4] = {kq.x, kq.y, kq.z, kq.w};
      const float vvals[4] = {vq.x, vq.y, vq.z, vq.w};
      // A element (K=l): lane-hi=(k>>3)&1, half=k>>4, idx=k&7
      const int hiA = (k >> 3) & 1, halfA = k >> 4, idxA = k & 7;
      // B element (K=l): lane-hi=(k>>3)&1, half=k>>4, idx=k&7
      const int hiB = (k >> 3) & 1, halfB = k >> 4, idxB = k & 7;
#pragma unroll
      for (int c = 0; c < 4; ++c) {
        const int d = col + c;
        const int ft = d >> 4;  // tile index along d (dt for A, et for B)
        const float kf = elup1(kvals[c] * psum - sMoff[d]) * m * SCALE;
        const int laneA = (d & 15) + (hiA << 4);
        sKfu[((((kk * 4 + ft) * 2 + halfA) * 32 + laneA) << 3) + idxA] =
            f32_to_bf16_bits(kf);
        const int laneB = (d & 15) + (hiB << 4);
        sVfu[((((kk * 4 + ft) * 2 + halfB) * 32 + laneB) << 3) + idxB] =
            f32_to_bf16_bits(vvals[c] * m);
      }
    }
    __syncthreads();
    // Prefetch next sub-chunk while this one is consumed (global_prefetch_b8).
    if (sc + 1 < NSUB) {
      const char* pk =
          (const char*)(K + ((size_t)bh * LL + l0 + SUB) * DD) + tid * 128;
      const char* pv =
          (const char*)(V + ((size_t)bh * LL + l0 + SUB) * DD) + tid * 128;
      __builtin_prefetch(pk, 0, 1);
      __builtin_prefetch(pv, 0, 1);
    }
    // KVtile += A(16x32 Kf^T) x B(32x16 Vm): 2 ds_load_b128 per operand.
#pragma unroll
    for (int kk = 0; kk < SUB / 32; ++kk) {
      const v16bf a = ldfrag(&sKf[(kk * 4 + dt) * 64], lane);
      const v16bf b0 = ldfrag(&sVf[(kk * 4 + et0) * 64], lane);
      const v16bf b1 = ldfrag(&sVf[(kk * 4 + et1) * 64], lane);
      acc0 = __builtin_amdgcn_wmma_f32_16x16x32_bf16(false, a, false, b0,
                                                     (short)0, acc0, false, false);
      acc1 = __builtin_amdgcn_wmma_f32_16x16x32_bf16(false, a, false, b1,
                                                     (short)0, acc1, false, false);
    }
  }

  // Merge partial tiles into KV[bh][64][64] (C layout: M = r + 8*hi, N = n0).
#pragma unroll
  for (int r = 0; r < 8; ++r) {
    const int M = r + (hi << 3);
    atomicAdd(&KV[((size_t)bh * DD + dt * 16 + M) * DD + et0 * 16 + n0], acc0[r]);
    atomicAdd(&KV[((size_t)bh * DD + dt * 16 + M) * DD + et1 * 16 + n0], acc1[r]);
  }
}

// -------- Pass 2: Out[l-tile] = Qf @ KV ------------------------------------
// sQf frag f=(rt*2+kk); sKVf frag g=(kk*4+ct); 64 v8bf each.
__global__ __launch_bounds__(256) void out_kernel(
    const float* __restrict__ Q, const float* __restrict__ KV,
    float* __restrict__ Out) {
  __shared__ v8bf sQf[16 * 64];  // 16 KB
  __shared__ v8bf sKVf[8 * 64];  // 8 KB
  unsigned short* sQfu = (unsigned short*)sQf;
  unsigned short* sKVfu = (unsigned short*)sKVf;

  const int bh = blockIdx.y;
  const int lbase = blockIdx.x * ROWS;
  const int tid = threadIdx.x;
  const int wave = tid >> 5;
  const int lane = tid & 31;
  const int hi = (lane >> 4) & 1;
  const int n0 = lane & 15;

  // Stage Qf rows as bf16 in A-fragment order (contiguous 4-ushort stores).
  for (int j = tid; j < ROWS * (DD / 4); j += 256) {
    const int row = j >> 4;
    const int col = (j & 15) * 4;
    const int l = lbase + row;
    const int rt = row >> 4, m = row & 15;
    const float4 q = *(const float4*)(Q + ((size_t)bh * LL + l) * DD + col);
    const float qvals[4] = {q.x, q.y, q.z, q.w};
    const int kkd = col >> 5;           // d-chunk
    const int dk = col & 31;            // K index within chunk (start)
    const int hiA = (dk >> 3) & 1, halfA = dk >> 4;
    const int laneA = m + (hiA << 4);
    unsigned short* qr =
        &sQfu[((((rt * 2 + kkd) * 2 + halfA) * 32 + laneA) << 3) + (dk & 7)];
#pragma unroll
    for (int c = 0; c < 4; ++c) qr[c] = f32_to_bf16_bits(elup1(qvals[c]) * SCALE);
  }
  // Stage this head's 64x64 KV as bf16 in B-fragment order.
  for (int j = tid; j < DD * (DD / 4); j += 256) {
    const int row = j >> 4;              // d (K dimension)
    const int col = (j & 15) * 4;        // e (N dimension)
    const float4 kv4 = *(const float4*)(KV + ((size_t)bh * DD + row) * DD + col);
    const float kvv[4] = {kv4.x, kv4.y, kv4.z, kv4.w};
    const int kkd = row >> 5;
    const int k = row & 31;
    const int hiB = (k >> 3) & 1, halfB = k >> 4, idxB = k & 7;
#pragma unroll
    for (int c = 0; c < 4; ++c) {
      const int e = col + c;
      const int ct = e >> 4;
      const int laneB = (e & 15) + (hiB << 4);
      sKVfu[((((kkd * 4 + ct) * 2 + halfB) * 32 + laneB) << 3) + idxB] =
          f32_to_bf16_bits(kvv[c]);
    }
  }
  __syncthreads();

  const int rt = wave;  // each wave owns one 16-row tile
  // A fragments are invariant across column tiles: load once.
  const v16bf a0 = ldfrag(&sQf[(rt * 2 + 0) * 64], lane);
  const v16bf a1 = ldfrag(&sQf[(rt * 2 + 1) * 64], lane);
#pragma unroll
  for (int ct = 0; ct < 4; ++ct) {
    const v16bf b0 = ldfrag(&sKVf[(0 * 4 + ct) * 64], lane);
    const v16bf b1 = ldfrag(&sKVf[(1 * 4 + ct) * 64], lane);
    v8f acc = {};
    acc = __builtin_amdgcn_wmma_f32_16x16x32_bf16(false, a0, false, b0,
                                                  (short)0, acc, false, false);
    acc = __builtin_amdgcn_wmma_f32_16x16x32_bf16(false, a1, false, b1,
                                                  (short)0, acc, false, false);
#pragma unroll
    for (int r = 0; r < 8; ++r) {
      const int l = lbase + rt * 16 + (hi << 3) + r;
      Out[((size_t)bh * LL + l) * DD + ct * 16 + n0] = acc[r];
    }
  }
}

extern "C" void kernel_launch(void* const* d_in, const int* in_sizes, int n_in,
                              void* d_out, int out_size, void* d_ws, size_t ws_size,
                              hipStream_t stream) {
  (void)in_sizes; (void)n_in; (void)out_size; (void)ws_size;
  const float* Q    = (const float*)d_in[0];
  const float* K    = (const float*)d_in[1];
  const float* V    = (const float*)d_in[2];
  const float* mask = (const float*)d_in[3];
  const float* pi   = (const float*)d_in[4];
  const float* mu   = (const float*)d_in[5];
  float* out = (float*)d_out;
  float* KV  = (float*)d_ws;  // B*H*64*64 f32 = 1 MB scratch

  const int nkv = BB * HH * DD * DD;
  zero_kv_kernel<<<(nkv + 255) / 256, 256, 0, stream>>>(KV, nkv);

  dim3 g1(LL / (SUB * NSUB), BB * HH);  // 16 x 64 blocks
  kv_kernel<<<g1, 256, 0, stream>>>(K, V, mask, pi, mu, KV);

  dim3 g2(LL / ROWS, BB * HH);          // 64 x 64 blocks
  out_kernel<<<g2, 256, 0, stream>>>(Q, KV, out);
}
